// AUAvULoss_6081673691429
// MI455X (gfx1250) — matrix operand order repaired and checked by reading, not compiled, not run on hardware
//
#include <hip/hip_runtime.h>
#include <hip/hip_fp16.h>

typedef __attribute__((ext_vector_type(16))) _Float16 v16h;
typedef __attribute__((ext_vector_type(8)))  float    v8f;

// ---------------------------------------------------------------------------
// ws layout (32-bit slots):
//  [0..83]  hist[t][n], t=0..20, n: 0=S_ac(le), 1=S_au(le), 2=S_ic(le), 3=S_iu(le)
//  [84] tot_au  [85] tot_iu  [86] focal_sum  [87] ce_sum
//  [88] umin_bits (uint)  [89] umax_bits (uint)  [90] label0 (int)
// ---------------------------------------------------------------------------

__device__ inline void row_stats(const float* __restrict__ pr, float* p,
                                 float& unc, float& conf, int& pred) {
  const float4* p4 = (const float4*)pr;
  float4 a = p4[0], b = p4[1];
  p[0]=a.x; p[1]=a.y; p[2]=a.z; p[3]=a.w; p[4]=b.x; p[5]=b.y; p[6]=b.z; p[7]=b.w;
  conf = p[0]; pred = 0;
  #pragma unroll
  for (int c = 1; c < 8; ++c) if (p[c] > conf) { conf = p[c]; pred = c; }
  unc = 0.0f;
  #pragma unroll
  for (int c = 0; c < 8; ++c) unc -= p[c] * logf(fmaxf(p[c], 1e-10f));
}

__device__ inline float wsumf(float v){ for(int o=16;o;o>>=1) v += __shfl_xor(v,o,32); return v; }
__device__ inline float wminf(float v){ for(int o=16;o;o>>=1) v = fminf(v,__shfl_xor(v,o,32)); return v; }
__device__ inline float wmaxf(float v){ for(int o=16;o;o>>=1) v = fmaxf(v,__shfl_xor(v,o,32)); return v; }

__global__ void avu_init(const float* __restrict__ y, float* wsf) {
  int t = threadIdx.x;
  if (t < 88) wsf[t] = 0.0f;
  unsigned* wsu = (unsigned*)wsf;
  if (t == 88) wsu[88] = 0x7F800000u;   // +inf bits (unc >= 0 -> unsigned order ok)
  if (t == 89) wsu[89] = 0u;
  if (t == 90) {                         // faithful bug: argmax of flattened one-hot
    float best = y[0]; int bi = 0;
    for (int c = 1; c < 8; ++c) { float v = y[c]; if (v > best) { best = v; bi = c; } }
    ((int*)wsf)[90] = bi;
  }
}

__global__ void avu_pass1(const float* __restrict__ probs, const float* __restrict__ yh,
                          const float* __restrict__ wts, float* wsf, long N) {
  const int label0 = ((const int*)wsf)[90];
  float focal = 0.f, ce = 0.f, tau = 0.f, tiu = 0.f;
  float umin = __int_as_float(0x7F800000), umax = 0.0f;
  long stride = (long)gridDim.x * blockDim.x;
  for (long s = (long)blockIdx.x * blockDim.x + threadIdx.x; s < N; s += stride) {
    float p[8], unc, conf; int pred;
    row_stats(probs + s * 8, p, unc, conf, pred);
    umin = fminf(umin, unc);
    umax = fmaxf(umax, unc);
    float T = tanhf(unc);
    bool corr = (pred == label0);
    tau += corr ? conf * T : 0.0f;
    tiu += corr ? 0.0f : (1.0f - conf) * T;
    const float4* y4 = (const float4*)(yh + s * 8);
    const float4* w4 = (const float4*)(wts + s * 8);
    float4 ya = y4[0], yb = y4[1], wa = w4[0], wb = w4[1];
    float yv[8] = {ya.x,ya.y,ya.z,ya.w,yb.x,yb.y,yb.z,yb.w};
    float wv[8] = {wa.x,wa.y,wa.z,wa.w,wb.x,wb.y,wb.z,wb.w};
    #pragma unroll
    for (int c = 0; c < 8; ++c) {
      float lp = logf(fmaxf(p[c], 1e-8f));
      focal -= yv[c] * lp * wv[c];
      ce    -= yv[c] * lp;
    }
  }
  focal = wsumf(focal); ce = wsumf(ce); tau = wsumf(tau); tiu = wsumf(tiu);
  umin = wminf(umin);   umax = wmaxf(umax);
  __shared__ float sm[8][6];
  int wid = threadIdx.x >> 5, lane = threadIdx.x & 31;
  if (lane == 0) { sm[wid][0]=focal; sm[wid][1]=ce; sm[wid][2]=tau;
                   sm[wid][3]=tiu;   sm[wid][4]=umin; sm[wid][5]=umax; }
  __syncthreads();
  if (threadIdx.x == 0) {
    float f=0.f, cc=0.f, a=0.f, iu=0.f, mn=__int_as_float(0x7F800000), mx=0.0f;
    int nw = blockDim.x >> 5;
    for (int w = 0; w < nw; ++w) {
      f += sm[w][0]; cc += sm[w][1]; a += sm[w][2]; iu += sm[w][3];
      mn = fminf(mn, sm[w][4]); mx = fmaxf(mx, sm[w][5]);
    }
    atomicAdd(&wsf[84], a);  atomicAdd(&wsf[85], iu);
    atomicAdd(&wsf[86], f);  atomicAdd(&wsf[87], cc);
    atomicMin((unsigned*)wsf + 88, __float_as_uint(mn));
    atomicMax((unsigned*)wsf + 89, __float_as_uint(mx));
  }
}

// pass2: WMMA contraction  S[t][n] = sum_k le[t,k] * w_n[k]
// A tile (16x32 f16): lane m=l%16 holds threshold row t; K(j)=j+(j>=8?8:0)+8*(l>=16)
// B tile (32x16 f16): lane holds column n=l%16;           K(j)=j+(j>=8?8:0)+8*(l>=16)
__global__ void avu_pass2(const float* __restrict__ probs, float* wsf, long N) {
  const int label0 = ((const int*)wsf)[90];
  const float umin = __uint_as_float(((const unsigned*)wsf)[88]);
  const float umax = __uint_as_float(((const unsigned*)wsf)[89]);
  const int lane = threadIdx.x & 31;
  const int g = lane >> 4;          // lane group
  const int m = lane & 15;          // A row (tile-local threshold) and B column
  const float span = umax - umin;
  const float th0 = umin + ((float)m / 20.0f) * span;                 // t = m (0..15)
  const int   t1i = 16 + m;
  const float th1 = (t1i <= 20) ? umin + ((float)t1i / 20.0f) * span
                                : __int_as_float(0x7F800000);         // dummy rows
  long wavesTotal = (long)gridDim.x * (blockDim.x >> 5);
  long gw = (long)blockIdx.x * (blockDim.x >> 5) + (threadIdx.x >> 5);
  v8f c0 = {}; v8f c1 = {};
  for (long base = gw * 32; base < N; base += wavesTotal * 32) {
    long s = base + lane;
    float unc = __int_as_float(0x7F800000), ps = 0.0f, T = 0.0f;
    if (s < N) {
      float p[8], conf; int pred;
      row_stats(probs + s * 8, p, unc, conf, pred);
      T = tanhf(unc);
      bool corr = (pred == label0);
      ps = corr ? conf : -(1.0f - conf);
    }
    unsigned pk = ((unsigned)__half_as_ushort(__float2half(T)) << 16)
                |  (unsigned)__half_as_ushort(__float2half(ps));
    v16h A0 = {}, A1 = {}, B = {};
    #pragma unroll
    for (int j = 0; j < 16; ++j) {
      int ka = j + ((j >= 8) ? 8 : 0) + (g << 3);
      float u = __shfl(unc, ka, 32);
      A0[j] = (_Float16)((u <= th0) ? 1.0f : 0.0f);
      A1[j] = (_Float16)((u <= th1) ? 1.0f : 0.0f);
      int kb = j + ((j >= 8) ? 8 : 0) + (g << 3);
      unsigned v = (unsigned)__shfl((int)pk, kb, 32);
      float psk = __half2float(__ushort_as_half((unsigned short)(v & 0xFFFFu)));
      float Tk  = __half2float(__ushort_as_half((unsigned short)(v >> 16)));
      float pss = (m < 2) ? psk : -psk;            // cols 0,1: correct; 2,3: incorrect
      float tw  = (m & 1) ? Tk : (1.0f - Tk);      // odd cols carry tanh factor
      float wv  = (m < 4) ? fmaxf(pss, 0.0f) * tw : 0.0f;
      B[j] = (_Float16)wv;
    }
    c0 = __builtin_amdgcn_wmma_f32_16x16x32_f16(false, A0, false, B, (short)0, c0, false, false);
    c1 = __builtin_amdgcn_wmma_f32_16x16x32_f16(false, A1, false, B, (short)0, c1, false, false);
  }
  // C layout: lane l, vgpr r -> N = l%16, M = r + 8*(l>=16)
  __shared__ float hist[84];
  for (int i = threadIdx.x; i < 84; i += blockDim.x) hist[i] = 0.0f;
  __syncthreads();
  if (m < 4) {
    #pragma unroll
    for (int r = 0; r < 8; ++r) {
      int t0 = r + (g << 3);                       // 0..15
      atomicAdd(&hist[t0 * 4 + m], c0[r]);
      int t1 = 16 + r + (g << 3);                  // 16..31, keep <21
      if (t1 < 21) atomicAdd(&hist[t1 * 4 + m], c1[r]);
    }
  }
  __syncthreads();
  for (int i = threadIdx.x; i < 84; i += blockDim.x) atomicAdd(&wsf[i], hist[i]);
}

__global__ void avu_finalize(const float* __restrict__ wsf, float* out, long N) {
  if (threadIdx.x != 0 || blockIdx.x != 0) return;
  float tot_au = wsf[84], tot_iu = wsf[85];
  float focal = wsf[86] / (float)N;
  float ce    = wsf[87] / (float)N;
  float avu[21];
  for (int t = 0; t < 21; ++t) {
    float n_ac = wsf[t * 4 + 0];
    float n_au = tot_au - wsf[t * 4 + 1];
    float n_ic = wsf[t * 4 + 2];
    float n_iu = tot_iu - wsf[t * 4 + 3];
    avu[t] = (n_ac + n_iu) / (n_ac + n_au + n_ic + n_iu + 1e-10f);
  }
  float auc = 0.0f;
  for (int t = 0; t < 20; ++t) {
    float dx = ((float)(t + 1) / 20.0f) - ((float)t / 20.0f);
    auc += 0.5f * (avu[t + 1] + avu[t]) * dx;
  }
  out[0] = -logf(auc + 1e-10f) + focal;   // BETA = 1.0
  out[1] = ce;
}

extern "C" void kernel_launch(void* const* d_in, const int* in_sizes, int n_in,
                              void* d_out, int out_size, void* d_ws, size_t ws_size,
                              hipStream_t stream) {
  const float* probs = (const float*)d_in[0];
  const float* y     = (const float*)d_in[1];
  const float* wts   = (const float*)d_in[2];
  long N = (long)in_sizes[0] / 8;
  float* wsf = (float*)d_ws;
  float* out = (float*)d_out;
  avu_init<<<1, 128, 0, stream>>>(y, wsf);
  avu_pass1<<<1024, 256, 0, stream>>>(probs, y, wts, wsf, N);
  avu_pass2<<<256, 256, 0, stream>>>(probs, wsf, N);
  avu_finalize<<<1, 32, 0, stream>>>(wsf, out, N);
}
